// RESK_40956808135036
// MI455X (gfx1250) — matrix operand order repaired and checked by reading, not compiled, not run on hardware
//
#include <hip/hip_runtime.h>
#include <hip/hip_bf16.h>

#define NNODES 100000
#define NEDGES 1600000

typedef __attribute__((ext_vector_type(16))) __bf16 v16bf;
typedef __attribute__((ext_vector_type(8)))  __bf16 v8bf;   // 16B, align 16
typedef __attribute__((ext_vector_type(8)))  float  v8f;

union FragBF {
  v16bf v;
  v8bf  h[2];
};

__device__ __forceinline__ unsigned pack_bf2(float lo, float hi) {
  union { __bf16 b[2]; unsigned u; } p;
  p.b[0] = (__bf16)lo;
  p.b[1] = (__bf16)hi;
  return p.u;          // lowers to v_cvt_pk_bf16_f32
}

// ---------------------------------------------------------------------------
// C[Nrows, NCOL] = A[Nrows, K] * W[K, NCOL]   (row-major f32 in/out)
// bf16 WMMA, f32 accumulate.  block = 128 threads = 4 waves.
// Block: 16 rows; wave w: columns [16w, 16w+16).  K % 32 == 0, NCOL <= 64.
//
// LDS layouts (pre-converted to bf16):
//   Wbf: column-major [n][k], row stride SK = K+8 bf16  -> each lane's
//        16-K B fragment is two 16B chunks 32B apart, conflict-free bank bases.
//   Abf: row-major [m][k within 32-step], stride 40 bf16 -> two 16B chunks
//        per lane fragment, conflict-free.
// ---------------------------------------------------------------------------
template <int K, int NCOL>
__launch_bounds__(128)
__global__ void gemm_wmma_bf16(const float* __restrict__ A,
                               const float* __restrict__ W,
                               float* __restrict__ C) {
  constexpr int SK   = K + 8;        // bf16 per Wbf row (pad for banks)
  constexpr int SKD  = SK / 2;       // dwords per Wbf row
  constexpr int ASTR = 40;           // bf16 per Abf row (80B, 16B aligned)

  __shared__ __bf16 Wbf[64 * SK];
  __shared__ __bf16 Abf[16 * ASTR];

  const int tid  = threadIdx.x;
  const int lane = tid & 31;
  const int wave = tid >> 5;
  const int rowBase = blockIdx.x << 4;
  const int colBase = wave << 4;

  // ---- Stage W -> bf16 column-major in LDS (once per block) ----
  {
    unsigned* wu = (unsigned*)Wbf;
    for (int i = tid; i < (K / 2) * NCOL; i += 128) {
      int n  = i % NCOL;
      int kp = i / NCOL;                         // k-pair index
      float w0 = W[(2 * kp) * NCOL + n];         // coalesced over n
      float w1 = W[(2 * kp + 1) * NCOL + n];
      wu[n * SKD + kp] = pack_bf2(w0, w1);
    }
  }

  const int m16   = lane & 15;
  const int hi    = lane >> 4;       // 0: lanes 0-15, 1: lanes 16-31
  const int kAoff = hi << 3;         // A frag K base: 0 or 8
  const int kBoff = hi << 4;         // 0 or 16 (unused; B frag uses kAoff)
  const int n     = colBase + m16;   // output column for this lane

  v8f acc = {};
  #pragma unroll
  for (int k0 = 0; k0 < K; k0 += 32) {
    __syncthreads();                 // frags of prev step consumed / W staged
    // ---- Stage 16x32 A tile -> bf16 in LDS (shared by all 4 waves) ----
    {
      unsigned* au = (unsigned*)Abf;
      for (int i = tid; i < 16 * 16; i += 128) {
        int r  = i >> 4;             // row 0..15
        int kp = i & 15;             // k-pair 0..15
        const float2 a2 = *reinterpret_cast<const float2*>(
            A + (long long)(rowBase + r) * K + k0 + 2 * kp);
        au[r * (ASTR / 2) + kp] = pack_bf2(a2.x, a2.y);
      }
    }
    __syncthreads();

    // A 16x32 bf16 fragment (ISA 7.12.2): K = kAoff+0..7 and kAoff+16..23.
    const __bf16* ap = Abf + m16 * ASTR + kAoff;
    FragBF a;
    a.h[0] = *reinterpret_cast<const v8bf*>(ap);
    a.h[1] = *reinterpret_cast<const v8bf*>(ap + 16);
    // B 32x16 bf16 fragment: K = kAoff+0..7 and kAoff+16..23 for column n.
    const __bf16* bp = Wbf + n * SK + k0 + kAoff;
    FragBF b;
    b.h[0] = *reinterpret_cast<const v8bf*>(bp);
    b.h[1] = *reinterpret_cast<const v8bf*>(bp + 16);

    acc = __builtin_amdgcn_wmma_f32_16x16x32_bf16(
        /*neg_a=*/false, a.v, /*neg_b=*/false, b.v,
        /*c_mod=*/(short)0, acc, /*reuse_a=*/false, /*reuse_b=*/false);
  }

  // C/D layout: VGPR j -> row M = j + 8*hi, column n.
  #pragma unroll
  for (int j = 0; j < 8; ++j) {
    int row = rowBase + j + (hi << 3);
    if (n < NCOL) C[(long long)row * NCOL + n] = acc[j];
  }
}

// ---------------------------------------------------------------------------
// agg[n][j] = bias[j]  (pre-load bias so edge scatter fuses the + b term)
// ---------------------------------------------------------------------------
__global__ void init_bias(float* __restrict__ agg, const float* __restrict__ b,
                          int dim) {
  long long i = (long long)blockIdx.x * blockDim.x + threadIdx.x;
  long long total = (long long)NNODES * dim;
  if (i < total) agg[i] = b[(int)(i % dim)];
}

// ---------------------------------------------------------------------------
// Edge scatter: agg[tgt[e]] += support[src[e]] * w[e]
// LANES threads per edge, each handles 4 contiguous columns: one float4
// gather (256B contiguous per edge for LANES=16) + 4 global_atomic_add_f32.
// ---------------------------------------------------------------------------
template <int LANES>
__launch_bounds__(256)
__global__ void spmm_scatter(const float* __restrict__ support,
                             const int* __restrict__ src,
                             const int* __restrict__ tgt,
                             const float* __restrict__ w,
                             float* __restrict__ agg) {
  constexpr int OUT = LANES * 4;
  long long gid = (long long)blockIdx.x * blockDim.x + threadIdx.x;
  long long e = gid / LANES;
  int l = (int)(gid - e * (long long)LANES);
  if (e >= NEDGES) return;
  int s = src[e];
  int t = tgt[e];
  float we = w[e];
  const float4 v =
      *reinterpret_cast<const float4*>(support + (long long)s * OUT + (l << 2));
  float* dst = agg + (long long)t * OUT + (l << 2);
  atomicAdd(dst + 0, v.x * we);
  atomicAdd(dst + 1, v.y * we);
  atomicAdd(dst + 2, v.z * we);
  atomicAdd(dst + 3, v.w * we);
}

// dst[i] = relu(agg[i])
__global__ void relu_k(float* __restrict__ dst, const float* __restrict__ agg,
                       long long total) {
  long long i = (long long)blockIdx.x * blockDim.x + threadIdx.x;
  if (i < total) dst[i] = fmaxf(agg[i], 0.0f);
}

// acc[i] = relu(acc[i]) + res[i]   (in place over the agg buffer)
__global__ void relu_res_k(float* __restrict__ acc, const float* __restrict__ res,
                           long long total) {
  long long i = (long long)blockIdx.x * blockDim.x + threadIdx.x;
  if (i < total) acc[i] = fmaxf(acc[i], 0.0f) + res[i];
}

// log_softmax over 40 columns, one thread per row (row lives in registers/L1)
__global__ void log_softmax40(const float* __restrict__ in,
                              float* __restrict__ out) {
  int r = blockIdx.x * blockDim.x + threadIdx.x;
  if (r >= NNODES) return;
  const float* p = in + (long long)r * 40;
  float m = -3.402823466e38f;
  #pragma unroll
  for (int j = 0; j < 40; ++j) m = fmaxf(m, p[j]);
  float s = 0.0f;
  #pragma unroll
  for (int j = 0; j < 40; ++j) s += __expf(p[j] - m);
  float ls = __logf(s) + m;
  float* o = out + (long long)r * 40;
  #pragma unroll
  for (int j = 0; j < 40; ++j) o[j] = p[j] - ls;
}

extern "C" void kernel_launch(void* const* d_in, const int* in_sizes, int n_in,
                              void* d_out, int out_size, void* d_ws, size_t ws_size,
                              hipStream_t stream) {
  (void)in_sizes; (void)n_in; (void)out_size; (void)ws_size;

  const float* x    = (const float*)d_in[0];
  const int*   src  = (const int*)d_in[1];
  const int*   tgt  = (const int*)d_in[2];
  const float* mw   = (const float*)d_in[3];
  const float* W0   = (const float*)d_in[4];
  const float* b0   = (const float*)d_in[5];
  const float* W1   = (const float*)d_in[6];
  const float* b1   = (const float*)d_in[7];
  const float* W2   = (const float*)d_in[8];
  const float* b2   = (const float*)d_in[9];
  const float* W3   = (const float*)d_in[10];
  const float* b3   = (const float*)d_in[11];
  float* out = (float*)d_out;

  // Three rotating N x 64 f32 buffers in workspace (76.8 MB total).
  float* S = (float*)d_ws;                       // support / scratch
  float* A = S + (size_t)NNODES * 64;            // agg (bias-preloaded)
  float* H = A + (size_t)NNODES * 64;            // current activations

  const long long tot64 = (long long)NNODES * 64;
  const long long tot40 = (long long)NNODES * 40;
  const dim3 gemmGrid(NNODES / 16), gemmBlk(128);
  const int gN64 = (int)((tot64 + 255) / 256);
  const int gN40 = (int)((tot40 + 255) / 256);
  const int gE64 = (int)(((long long)NEDGES * 16 + 255) / 256);
  const int gE40 = (int)(((long long)NEDGES * 10 + 255) / 256);

  // ---- Layer 0: h0 = relu(gc(x, W0, b0)) ----
  gemm_wmma_bf16<128, 64><<<gemmGrid, gemmBlk, 0, stream>>>(x, W0, S);
  init_bias<<<gN64, 256, 0, stream>>>(A, b0, 64);
  spmm_scatter<16><<<gE64, 256, 0, stream>>>(S, src, tgt, mw, A);
  relu_k<<<gN64, 256, 0, stream>>>(H, A, tot64);

  // ---- Layer 1 (residual): h1 = relu(gc(h0, W1, b1)) + h0 ----
  gemm_wmma_bf16<64, 64><<<gemmGrid, gemmBlk, 0, stream>>>(H, W1, S);
  init_bias<<<gN64, 256, 0, stream>>>(A, b1, 64);
  spmm_scatter<16><<<gE64, 256, 0, stream>>>(S, src, tgt, mw, A);
  relu_res_k<<<gN64, 256, 0, stream>>>(A, H, tot64);
  { float* t = A; A = H; H = t; }   // H = h1, A = free scratch

  // ---- Layer 2 (residual): h2 = relu(gc(h1, W2, b2)) + h1 ----
  gemm_wmma_bf16<64, 64><<<gemmGrid, gemmBlk, 0, stream>>>(H, W2, S);
  init_bias<<<gN64, 256, 0, stream>>>(A, b2, 64);
  spmm_scatter<16><<<gE64, 256, 0, stream>>>(S, src, tgt, mw, A);
  relu_res_k<<<gN64, 256, 0, stream>>>(A, H, tot64);
  { float* t = A; A = H; H = t; }   // H = h2

  // ---- Layer 3: out = log_softmax(gc(h2, W3, b3)) ----
  gemm_wmma_bf16<64, 40><<<gemmGrid, gemmBlk, 0, stream>>>(H, W3, S);
  init_bias<<<gN40, 256, 0, stream>>>(A, b3, 40);
  spmm_scatter<10><<<gE40, 256, 0, stream>>>(S, src, tgt, mw, A);
  log_softmax40<<<(NNODES + 255) / 256, 256, 0, stream>>>(A, out);
}
